// KagomeConv2D_20633022890671
// MI455X (gfx1250) — matrix operand order repaired and checked by reading, not compiled, not run on hardware
//
#include <hip/hip_runtime.h>

// ---------------------------------------------------------------------------
// KagomeConv2D on MI455X (gfx1250): implicit-GEMM, bf16 hi/lo split WMMA,
// double-buffered LDS software pipeline.
// x:[2048,256,8,8] f32, w:[256,256,4,4] f32 x3, b:[256] f32 x3 -> out f32
// ---------------------------------------------------------------------------

typedef __attribute__((ext_vector_type(16))) __bf16 v16bf;
typedef __attribute__((ext_vector_type(8)))  float  v8f;

#define XSB   16384u  // x batch stride (256*64)
#define WSO   4096u   // w cout stride (256*16)
#define OSB   16384u  // out batch stride

// padded 10x10 cell -> offset into 8x8 spatial (r*8+c) of original x, or -1 == zero.
// Encodes zero-pad + the 18 boundary-fixup copies (sources are original interior).
__constant__ signed char padMap[10][10] = {
  {59,60,61,62,-1,-1,-1,-1,-1,-1},
  {-1, 0, 1, 2, 3,32, 5, 6, 7,-1},
  {-1, 8, 9,10,11,12,41,14,15,-1},
  {-1,16,17,18,19,20,21,50,23,-1},
  {-1,24,25,26,27,28,29,30,59,-1},
  {-1,32,33,34,35,36,37,38,39,-1},
  {47,40,41,42,43,44,45,46,47, 8},
  {-1,48,49,50,51,52,53,54,55,16},
  {63,56,57,58,59,60,61,62,63,24},
  {-1,12,-1,30,-1, 0,-1, 2,-1,32},
};

// Mask-nonzero taps per conv type (up/left/right).
__constant__ unsigned char tapKR[3][5] = {{0,0,1,2,2},{1,2,2,2,3},{1,2,2,2,3}};
__constant__ unsigned char tapKC[3][5] = {{0,1,1,1,2},{1,0,1,2,1},{1,1,2,3,3}};

// Active (conv,r,c) positions (48 minus the 12 scatter-zeroed ones) + output cell.
struct Pos { unsigned char t, r, c, cell; };
__constant__ Pos g_pos[36] = {
  {0,0,0, 0},{0,0,1, 2},{0,1,0,16},{0,1,1,18},{0,1,2,20},{0,2,0,32},
  {0,2,1,34},{0,2,2,36},{0,2,3,38},{0,3,1,50},{0,3,2,52},{0,3,3,54},
  {1,0,0, 8},{1,0,1,10},{1,0,2,12},{1,1,0,24},{1,1,1,26},{1,1,2,28},
  {1,1,3,30},{1,2,1,42},{1,2,2,44},{1,2,3,46},{1,3,2,60},{1,3,3,62},
  {2,0,0, 9},{2,0,1,11},{2,1,0,25},{2,1,1,27},{2,1,2,29},{2,2,0,41},
  {2,2,1,43},{2,2,2,45},{2,2,3,47},{2,3,1,59},{2,3,2,61},{2,3,3,63},
};

union FragU { uint4 q[2]; v16bf v; };

// A fragment (16x32 bf16): lane half h, row m. K(e) = e + 8h + 8*(e>=8)
__device__ inline v16bf loadFragA(const __bf16* plane, int row, int half) {
  FragU f;
  const char* p = (const char*)plane + row * 80 + half * 16;
  f.q[0] = *(const uint4*)(p);
  f.q[1] = *(const uint4*)(p + 32);
  return f.v;
}
// B fragment (32x16 bf16): lane half h, row n. K(e) = e + 8h + 8*(e>=8)
__device__ inline v16bf loadFragB(const __bf16* plane, int row, int half) {
  FragU f;
  const char* p = (const char*)plane + row * 80 + half * 16;
  f.q[0] = *(const uint4*)(p);
  f.q[1] = *(const uint4*)(p + 32);
  return f.v;
}

__device__ inline unsigned pack2(__bf16 a, __bf16 b) {
  union { __bf16 h[2]; unsigned u; } u;
  u.h[0] = a; u.h[1] = b; return u.u;
}

__global__ void __launch_bounds__(256)
kagome_zero(float4* __restrict__ out) {
  out[blockIdx.x * 256u + threadIdx.x] = make_float4(0.f, 0.f, 0.f, 0.f);
}

__global__ void __launch_bounds__(256)
kagome_gemm(const float* __restrict__ x,
            const float* __restrict__ w0, const float* __restrict__ bias0,
            const float* __restrict__ w1, const float* __restrict__ bias1,
            const float* __restrict__ w2, const float* __restrict__ bias2,
            float* __restrict__ out) {
  // ---- uniform (scalar) setup -------------------------------------------
  const Pos P = g_pos[blockIdx.y];
  const float* wgt  = (P.t == 0) ? w0 : (P.t == 1) ? w1 : w2;
  const float* bias = (P.t == 0) ? bias0 : (P.t == 1) ? bias1 : bias2;

  int toff[5], twi[5], nv = 0;
  for (int j = 0; j < 5; ++j) {
    int kr = tapKR[P.t][j], kc = tapKC[P.t][j];
    int off = padMap[2 * P.r + kr][2 * P.c + kc];
    if (off >= 0) { toff[nv] = off; twi[nv] = kr * 4 + kc; ++nv; }
  }
  const int cell  = P.cell;
  const int total = nv * 8;             // K chunks of 32 channels

  const int nT = blockIdx.x & 1;        // 2 N tiles of 128
  const int mT = blockIdx.x >> 1;       // 16 M tiles of 128
  const int b0 = mT * 128;
  const int n0 = nT * 128;

  // ---- LDS: double-buffered bf16 hi/lo planes, 80B pitch ----------------
  __shared__ __align__(16) __bf16 sA[2][2][128][40];   // [buf][plane][m][k]
  __shared__ __align__(16) __bf16 sB[2][2][128][40];   // [buf][plane][n][k]

  const int tid   = threadIdx.x;
  const int lane  = tid & 31;
  const int half  = lane >> 4;
  const int l15   = lane & 15;
  const int wv    = tid >> 5;           // 8 waves: 4 (M) x 2 (N)
  const int wm    = wv & 3;             // 32-row M strip
  const int wn    = wv >> 2;            // 64-col N strip

  // Loader mapping: each thread owns channel pair (kk0,kk0+1) x 8 rows.
  const int kpair = tid & 15;
  const int kk0   = kpair * 2;
  const int grp   = tid >> 4;           // 16 row groups of 8

  v8f acc[2][4] = {};                   // 2x4 subtiles of 16x16 per wave

  float aR[16], bR[16];

  auto fetch = [&](int kc) {
    const int kt = kc >> 3, cb = kc & 7;
    const int off = toff[kt], tap = twi[kt];
    const int ch0 = cb * 32;
    const float* xp = x + (size_t)(b0 + grp * 8) * XSB
                        + (size_t)(ch0 + kk0) * 64u + off;
    const float* wp = wgt + (size_t)(n0 + grp * 8) * WSO
                          + (size_t)(ch0 + kk0) * 16u + tap;
    #pragma unroll
    for (int it = 0; it < 8; ++it) {
      aR[2 * it]     = xp[(size_t)it * XSB];
      aR[2 * it + 1] = xp[(size_t)it * XSB + 64u];
      bR[2 * it]     = wp[(size_t)it * WSO];
      bR[2 * it + 1] = wp[(size_t)it * WSO + 16u];
    }
  };

  auto stage = [&](int buf) {
    #pragma unroll
    for (int it = 0; it < 8; ++it) {
      const int row = grp * 8 + it;
      float a0 = aR[2 * it], a1 = aR[2 * it + 1];
      __bf16 ah0 = (__bf16)a0, ah1 = (__bf16)a1;
      __bf16 al0 = (__bf16)(a0 - (float)ah0), al1 = (__bf16)(a1 - (float)ah1);
      *(unsigned*)((char*)&sA[buf][0][row][0] + kpair * 4) = pack2(ah0, ah1);
      *(unsigned*)((char*)&sA[buf][1][row][0] + kpair * 4) = pack2(al0, al1);
      float b0v = bR[2 * it], b1v = bR[2 * it + 1];
      __bf16 bh0 = (__bf16)b0v, bh1 = (__bf16)b1v;
      __bf16 bl0 = (__bf16)(b0v - (float)bh0), bl1 = (__bf16)(b1v - (float)bh1);
      *(unsigned*)((char*)&sB[buf][0][row][0] + kpair * 4) = pack2(bh0, bh1);
      *(unsigned*)((char*)&sB[buf][1][row][0] + kpair * 4) = pack2(bl0, bl1);
    }
  };

  // ---- software pipeline ------------------------------------------------
  fetch(0);
  stage(0);
  __syncthreads();

  for (int kc = 0; kc < total; ++kc) {
    const int buf = kc & 1;
    if (kc + 1 < total) fetch(kc + 1);          // overlap with WMMAs below
    if (kc + 2 < total) {                       // prefetch chunk kc+2
      const int kt = (kc + 2) >> 3, cb = (kc + 2) & 7;
      __builtin_prefetch(x + (size_t)(b0 + grp * 8) * XSB
                           + (size_t)(cb * 32 + kk0) * 64u + toff[kt], 0, 3);
    }

    v16bf aH[2], aL[2];
    #pragma unroll
    for (int i = 0; i < 2; ++i) {
      int row = wm * 32 + i * 16 + l15;
      aH[i] = loadFragA(&sA[buf][0][0][0], row, half);
      aL[i] = loadFragA(&sA[buf][1][0][0], row, half);
    }
    #pragma unroll
    for (int j = 0; j < 4; ++j) {
      int row = wn * 64 + j * 16 + l15;
      v16bf bH = loadFragB(&sB[buf][0][0][0], row, half);
      v16bf bL = loadFragB(&sB[buf][1][0][0], row, half);
      #pragma unroll
      for (int i = 0; i < 2; ++i) {
        acc[i][j] = __builtin_amdgcn_wmma_f32_16x16x32_bf16(
            false, aH[i], false, bH, (short)0, acc[i][j], false, false);
        acc[i][j] = __builtin_amdgcn_wmma_f32_16x16x32_bf16(
            false, aH[i], false, bL, (short)0, acc[i][j], false, false);
        acc[i][j] = __builtin_amdgcn_wmma_f32_16x16x32_bf16(
            false, aL[i], false, bH, (short)0, acc[i][j], false, false);
      }
    }

    if (kc + 1 < total) {
      stage((kc + 1) & 1);
      __syncthreads();
    }
  }

  // ---- epilogue: bias + scatter into interleaved output cell -------------
  float bv[4];
  #pragma unroll
  for (int j = 0; j < 4; ++j) bv[j] = bias[n0 + wn * 64 + j * 16 + l15];

  #pragma unroll
  for (int i = 0; i < 2; ++i) {
    #pragma unroll
    for (int j = 0; j < 4; ++j) {
      const int n = n0 + wn * 64 + j * 16 + l15;       // cout (col = lane&15)
      #pragma unroll
      for (int v = 0; v < 8; ++v) {
        const int b = b0 + wm * 32 + i * 16 + v + 8 * half;  // M = vgpr + 8*half
        out[(size_t)b * OSB + (size_t)n * 64u + cell] = acc[i][j][v] + bv[j];
      }
    }
  }
}

extern "C" void kernel_launch(void* const* d_in, const int* in_sizes, int n_in,
                              void* d_out, int out_size, void* d_ws, size_t ws_size,
                              hipStream_t stream) {
  (void)in_sizes; (void)n_in; (void)d_ws; (void)ws_size; (void)out_size;
  const float* x      = (const float*)d_in[0];
  const float* w_up   = (const float*)d_in[1];
  const float* b_up   = (const float*)d_in[2];
  const float* w_left = (const float*)d_in[3];
  const float* b_left = (const float*)d_in[4];
  const float* w_rt   = (const float*)d_in[5];
  const float* b_rt   = (const float*)d_in[6];
  float* out = (float*)d_out;

  // 1) zero the full output (covers interleave zeros + ZR/ZC scatter-zeros)
  kagome_zero<<<32768, 256, 0, stream>>>((float4*)out);

  // 2) 36 implicit GEMMs (one per surviving conv output position)
  dim3 grid(32 /* 16 Mtiles x 2 Ntiles */, 36);
  kagome_gemm<<<grid, 256, 0, stream>>>(x, w_up, b_up, w_left, b_left,
                                        w_rt, b_rt, out);
}
